// GraphTransformer_63969242906717
// MI455X (gfx1250) — compile-verified, matrix operand order checked by reading, not run on hardware
//
#include <hip/hip_runtime.h>
#include <hip/hip_bf16.h>
#include <math.h>

// Problem constants (match reference)
#define NNODE 8192
#define BGR   64
#define PNODE 128
#define DDIM  512
#define EEDGE 131072
#define LLAYER 4
#define HGH   8
#define CGD   64
#define HDD   64
#define OUTD  128
#define NFF   9
#define ATOMV 64
#define EFF   3
#define BONDV 8
#define ETOT  (EEDGE + NNODE)   // edges + self loops

typedef __bf16 bf16_t;
typedef __attribute__((ext_vector_type(16))) __bf16 v16bf;
typedef __attribute__((ext_vector_type(8)))  __bf16 v8bf;
typedef __attribute__((ext_vector_type(8)))  float  v8f;

// ---------- helpers ----------
__device__ __forceinline__ bf16_t f2bf(float f) {
  unsigned u = __float_as_uint(f);
  u += 0x7FFFu + ((u >> 16) & 1u);          // round-to-nearest-even
  unsigned short h = (unsigned short)(u >> 16);
  bf16_t b;
  __builtin_memcpy(&b, &h, 2);
  return b;
}

// ordered-uint encoding so atomicMax(uint) == max over floats
__device__ __forceinline__ unsigned fenc(float f) {
  unsigned u = __float_as_uint(f);
  return (u & 0x80000000u) ? ~u : (u | 0x80000000u);
}
__device__ __forceinline__ float fdec(unsigned u) {
  unsigned v = (u & 0x80000000u) ? (u & 0x7FFFFFFFu) : ~u;
  return __uint_as_float(v);
}

// Load a 16x32 (or 32x16) bf16 WMMA fragment for one lane from a 32-elem K-chunk
// CDNA5 16-bit layout: lane half lh: elems 0..7 -> K=lh*8+0..7, 8..15 -> K=16+lh*8+0..7
__device__ __forceinline__ v16bf frag_from(const bf16_t* chunk, int lh) {
  const v8bf lo = *(const v8bf*)(chunk + lh * 8);
  const v8bf hi = *(const v8bf*)(chunk + 16 + lh * 8);
  v16bf r;
#pragma unroll
  for (int i = 0; i < 8; i++) { r[i] = lo[i]; r[i + 8] = hi[i]; }
  return r;
}

__device__ __forceinline__ v8f wmma_bf(v16bf a, v16bf b, v8f c) {
  return __builtin_amdgcn_wmma_f32_16x16x32_bf16(false, a, false, b, (short)0, c,
                                                 false, false);
}

// ---------- generic bf16 WMMA GEMM: C[M,N] = A[M,K] @ B[K,N] (+bias, relu, bf16 copy)
// M % 128 == 0, N % 128 == 0, K % 32 == 0
__global__ __launch_bounds__(256) void gemm_bf16_kernel(
    const bf16_t* __restrict__ A, const bf16_t* __restrict__ B,
    const float* __restrict__ bias, float* __restrict__ C,
    bf16_t* __restrict__ Cbf, int M, int N, int K, int relu)
{
  __shared__ bf16_t sa[128 * 32];
  __shared__ bf16_t sbT[128 * 32];  // B tile stored N-major (transposed)
  const int tid = threadIdx.x;
  const int wid = tid >> 5, lane = tid & 31;
  const int lm = lane & 15, lh = lane >> 4;
  const int m0 = blockIdx.y * 128, n0 = blockIdx.x * 128;
  const int wm = wid & 3, wn = wid >> 2;     // 4(M) x 2(N) waves
  const int mbase = wm * 32, nbase = wn * 64;
  v8f acc[2][4] = {};

  const int arow = tid >> 1, ahalf = (tid & 1) * 16;   // A: 128 rows x 32
  const int bkk = tid >> 3, bns = (tid & 7) * 16;      // B: 32 rows x 128

  for (int k0 = 0; k0 < K; k0 += 32) {
    {
      const bf16_t* g = A + (size_t)(m0 + arow) * K + k0 + ahalf;
      v8bf p0 = *(const v8bf*)g;
      v8bf p1 = *(const v8bf*)(g + 8);
      *(v8bf*)&sa[arow * 32 + ahalf] = p0;
      *(v8bf*)&sa[arow * 32 + ahalf + 8] = p1;
    }
    {
      const bf16_t* g = B + (size_t)(k0 + bkk) * N + n0 + bns;
      v8bf p0 = *(const v8bf*)g;
      v8bf p1 = *(const v8bf*)(g + 8);
#pragma unroll
      for (int j = 0; j < 8; j++) sbT[(bns + j) * 32 + bkk] = p0[j];
#pragma unroll
      for (int j = 0; j < 8; j++) sbT[(bns + 8 + j) * 32 + bkk] = p1[j];
    }
    __syncthreads();
    v16bf af[2], bf[4];
#pragma unroll
    for (int i = 0; i < 2; i++)
      af[i] = frag_from(&sa[(mbase + i * 16 + lm) * 32], lh);
#pragma unroll
    for (int j = 0; j < 4; j++)
      bf[j] = frag_from(&sbT[(nbase + j * 16 + lm) * 32], lh);
#pragma unroll
    for (int i = 0; i < 2; i++)
#pragma unroll
      for (int j = 0; j < 4; j++) acc[i][j] = wmma_bf(af[i], bf[j], acc[i][j]);
    __syncthreads();
  }

#pragma unroll
  for (int i = 0; i < 2; i++) {
#pragma unroll
    for (int j = 0; j < 4; j++) {
      const int nn = n0 + nbase + j * 16 + lm;
      const float bv = bias ? bias[nn] : 0.f;
#pragma unroll
      for (int r = 0; r < 8; r++) {
        const int mm = m0 + mbase + i * 16 + r + lh * 8;
        float v = acc[i][j][r] + bv;
        if (relu) v = fmaxf(v, 0.f);
        C[(size_t)mm * N + nn] = v;
        if (Cbf) Cbf[(size_t)mm * N + nn] = f2bf(v);
      }
    }
  }
}

// ---------- dense per-graph attention, WMMA QK^T + softmax + WMMA PV ----------
// grid: (B, HT, P/32); block: 256
__global__ __launch_bounds__(256) void attn_kernel(
    const float* __restrict__ qkv, float* __restrict__ o, bf16_t* __restrict__ obf)
{
  __shared__ bf16_t qsd[32 * 64];
  __shared__ bf16_t kld[128 * 64];
  __shared__ bf16_t vtd[64 * 128];   // V transposed: vtd[d][p]
  __shared__ float  sc[32 * 128];
  __shared__ bf16_t prob[32 * 128];
  __shared__ float  rsum[32];
  const int b = blockIdx.x, h = blockIdx.y, qs = blockIdx.z;
  const int tid = threadIdx.x;
  const size_t gb = (size_t)(b * PNODE) * (3 * DDIM);

  for (int i = tid; i < 128 * 64; i += 256) {
    int p = i >> 6, d = i & 63;
    size_t rb = gb + (size_t)p * (3 * DDIM) + h * HDD + d;
    kld[p * 64 + d] = f2bf(qkv[rb + DDIM]);
    vtd[d * 128 + p] = f2bf(qkv[rb + 2 * DDIM]);
  }
  for (int i = tid; i < 32 * 64; i += 256) {
    int p = i >> 6, d = i & 63;
    qsd[p * 64 + d] =
        f2bf(qkv[gb + (size_t)(qs * 32 + p) * (3 * DDIM) + h * HDD + d]);
  }
  __syncthreads();

  const int wid = tid >> 5, lane = tid & 31, lm = lane & 15, lh = lane >> 4;
  {  // scores strip [32 x 128]: wave -> mt = w&1, ntiles {2*(w>>1), 2*(w>>1)+1}
    const int mt = wid & 1, nt0 = (wid >> 1) * 2;
    v8f a0 = {}, a1 = {};
#pragma unroll
    for (int ks = 0; ks < 2; ks++) {
      v16bf aa = frag_from(&qsd[(mt * 16 + lm) * 64 + ks * 32], lh);
      v16bf b0 = frag_from(&kld[(nt0 * 16 + lm) * 64 + ks * 32], lh);
      v16bf b1 = frag_from(&kld[((nt0 + 1) * 16 + lm) * 64 + ks * 32], lh);
      a0 = wmma_bf(aa, b0, a0);
      a1 = wmma_bf(aa, b1, a1);
    }
#pragma unroll
    for (int r = 0; r < 8; r++) {
      int mrow = mt * 16 + r + lh * 8;
      sc[mrow * 128 + nt0 * 16 + lm] = a0[r] * 0.125f;        // 1/sqrt(64)
      sc[mrow * 128 + (nt0 + 1) * 16 + lm] = a1[r] * 0.125f;
    }
  }
  __syncthreads();
  if (tid < 32) {  // softmax row tid (unnormalized exp; scale at output)
    float mx = -3.4e38f;
    for (int c = 0; c < 128; c++) mx = fmaxf(mx, sc[tid * 128 + c]);
    float s = 0.f;
    for (int c = 0; c < 128; c++) {
      float e = expf(sc[tid * 128 + c] - mx);
      s += e;
      prob[tid * 128 + c] = f2bf(e);
    }
    rsum[tid] = s;
  }
  __syncthreads();
  {  // o strip = P @ V : 2 x 4 tiles; wave -> mt = w&1, nt = w>>1
    const int mt = wid & 1, nt = wid >> 1;
    v8f acc = {};
#pragma unroll
    for (int ks = 0; ks < 4; ks++) {
      v16bf aa = frag_from(&prob[(mt * 16 + lm) * 128 + ks * 32], lh);
      v16bf bb = frag_from(&vtd[(nt * 16 + lm) * 128 + ks * 32], lh);
      acc = wmma_bf(aa, bb, acc);
    }
#pragma unroll
    for (int r = 0; r < 8; r++) {
      int mloc = mt * 16 + r + lh * 8;
      int mg = b * PNODE + qs * 32 + mloc;
      int ng = h * HDD + nt * 16 + lm;
      float v = acc[r] / rsum[mloc];
      o[(size_t)mg * DDIM + ng] = v;
      obf[(size_t)mg * DDIM + ng] = f2bf(v);
    }
  }
}

// ---------- small kernels ----------
__global__ void cvt_bf16_kernel(const float* __restrict__ in,
                                bf16_t* __restrict__ out, size_t n) {
  size_t i = (size_t)blockIdx.x * blockDim.x + threadIdx.x;
  if (i < n) out[i] = f2bf(in[i]);
}

// W_eff[l*8+h][k] = sum_c gat_We[l][k][h*64+c] * att_edge[l][h][c]
__global__ void weff_kernel(const float* __restrict__ gat_We,
                            const float* __restrict__ att_edge,
                            float* __restrict__ weff) {
  int idx = blockIdx.x * blockDim.x + threadIdx.x;
  if (idx >= LLAYER * HGH * DDIM) return;
  int l = idx / (HGH * DDIM);
  int h = (idx / DDIM) % HGH;
  int k = idx % DDIM;
  const float* wcol = gat_We + ((size_t)(l * DDIM + k)) * DDIM + h * CGD;
  const float* ae = att_edge + (size_t)(l * HGH + h) * CGD;
  float s = 0.f;
  for (int c = 0; c < CGD; c++) s += wcol[c] * ae[c];
  weff[(size_t)(l * HGH + h) * DDIM + k] = s;
}

__global__ void atom_embed_kernel(const int* __restrict__ xf,
                                  const float* __restrict__ emb,
                                  float* __restrict__ x, bf16_t* __restrict__ xbf) {
  int idx = blockIdx.x * blockDim.x + threadIdx.x;
  if (idx >= NNODE * DDIM) return;
  int n = idx >> 9, d = idx & 511;
  float s = 0.f;
#pragma unroll
  for (int f = 0; f < NFF; f++)
    s += emb[((size_t)(f * ATOMV + xf[n * NFF + f])) * DDIM + d];
  x[idx] = s;
  xbf[idx] = f2bf(s);
}

__global__ void deg_kernel(const int* __restrict__ dst, float* __restrict__ deg) {
  int e = blockIdx.x * blockDim.x + threadIdx.x;
  if (e < EEDGE) atomicAdd(&deg[dst[e]], 1.0f);
}

__global__ void bond_mean_kernel(const int* __restrict__ eaf,
                                 const int* __restrict__ dst,
                                 const float* __restrict__ emb,
                                 float* __restrict__ macc) {
  size_t idx = (size_t)blockIdx.x * blockDim.x + threadIdx.x;
  if (idx >= (size_t)EEDGE * DDIM) return;
  int e = (int)(idx >> 9), d = (int)(idx & 511);
  float v = 0.f;
#pragma unroll
  for (int f = 0; f < EFF; f++)
    v += emb[((size_t)(f * BONDV + eaf[e * EFF + f])) * DDIM + d];
  atomicAdd(&macc[(size_t)dst[e] * DDIM + d], v);
}

__global__ void mean_fin_kernel(float* __restrict__ me, const float* __restrict__ deg) {
  int idx = blockIdx.x * blockDim.x + threadIdx.x;
  if (idx >= NNODE * DDIM) return;
  me[idx] /= fmaxf(deg[idx >> 9], 1.0f);
}

// a_e_all[r][0..31] = e_a_row(r) @ W_eff^T  (one block per row)
__global__ __launch_bounds__(256) void aeall_kernel(
    const int* __restrict__ eaf, const float* __restrict__ emb,
    const float* __restrict__ mean_e, const float* __restrict__ weff,
    float* __restrict__ a_e_all) {
  __shared__ float erow[DDIM];
  __shared__ float part[32][8];
  const int r = blockIdx.x, tid = threadIdx.x;
  for (int d = tid; d < DDIM; d += 256) {
    float v;
    if (r < EEDGE) {
      v = emb[((size_t)(0 * BONDV + eaf[r * EFF + 0])) * DDIM + d] +
          emb[((size_t)(1 * BONDV + eaf[r * EFF + 1])) * DDIM + d] +
          emb[((size_t)(2 * BONDV + eaf[r * EFF + 2])) * DDIM + d];
    } else {
      v = mean_e[(size_t)(r - EEDGE) * DDIM + d];
    }
    erow[d] = v;
  }
  __syncthreads();
  const int o = tid & 31, ch = tid >> 5;  // 32 outputs x 8 chunks of 64
  const float* wr = weff + (size_t)o * DDIM + ch * 64;
  const float* er = erow + ch * 64;
  float p = 0.f;
#pragma unroll
  for (int i = 0; i < 64; i++) p += er[i] * wr[i];
  part[o][ch] = p;
  __syncthreads();
  if (tid < 32) {
    float s = 0.f;
#pragma unroll
    for (int j = 0; j < 8; j++) s += part[tid][j];
    a_e_all[(size_t)r * 32 + tid] = s;
  }
}

// a_s/a_d per (node, head) from xh
__global__ void asd_kernel(const float* __restrict__ xh,
                           const float* __restrict__ att_src_l,
                           const float* __restrict__ att_dst_l,
                           float* __restrict__ a_s, float* __restrict__ a_d) {
  int idx = blockIdx.x * blockDim.x + threadIdx.x;
  if (idx >= NNODE * HGH) return;
  int n = idx >> 3, h = idx & 7;
  const float* xr = xh + (size_t)n * DDIM + h * CGD;
  const float* as = att_src_l + h * CGD;
  const float* ad = att_dst_l + h * CGD;
  float s = 0.f, d = 0.f;
#pragma unroll 8
  for (int c = 0; c < CGD; c++) { s += xr[c] * as[c]; d += xr[c] * ad[c]; }
  a_s[idx] = s;
  a_d[idx] = d;
}

__global__ void logits_kernel(const int* __restrict__ src, const int* __restrict__ dst,
                              const float* __restrict__ a_s, const float* __restrict__ a_d,
                              const float* __restrict__ a_e_all, int layer,
                              float* __restrict__ lg, unsigned* __restrict__ mmax) {
  int idx = blockIdx.x * blockDim.x + threadIdx.x;
  if (idx >= ETOT * HGH) return;
  int r = idx >> 3, h = idx & 7;
  int s, dn;
  if (r < EEDGE) { s = src[r]; dn = dst[r]; } else { s = dn = r - EEDGE; }
  float v = a_s[s * HGH + h] + a_d[dn * HGH + h] +
            a_e_all[(size_t)r * 32 + layer * HGH + h];
  v = (v > 0.f) ? v : 0.2f * v;   // leaky relu
  lg[idx] = v;
  atomicMax(&mmax[dn * HGH + h], fenc(v));
}

__global__ void pexp_kernel(const int* __restrict__ dst,
                            float* __restrict__ lg, const unsigned* __restrict__ mmax,
                            float* __restrict__ den) {
  int idx = blockIdx.x * blockDim.x + threadIdx.x;
  if (idx >= ETOT * HGH) return;
  int r = idx >> 3, h = idx & 7;
  int dn = (r < EEDGE) ? dst[r] : (r - EEDGE);
  float p = expf(lg[idx] - fdec(mmax[dn * HGH + h]));
  lg[idx] = p;
  atomicAdd(&den[dn * HGH + h], p);
}

// xg[dst] += alpha_h * xh[src]; one block per edge row, 512 threads over D
__global__ __launch_bounds__(512) void agg_kernel(
    const int* __restrict__ src, const int* __restrict__ dst,
    const float* __restrict__ lg, const float* __restrict__ den,
    const float* __restrict__ xh, float* __restrict__ xg) {
  __shared__ float alpha[HGH];
  __shared__ int ss, sd;
  const int r = blockIdx.x, tid = threadIdx.x;
  if (tid == 0) {
    if (r < EEDGE) { ss = src[r]; sd = dst[r]; } else { ss = sd = r - EEDGE; }
  }
  __syncthreads();
  if (tid < HGH) alpha[tid] = lg[(size_t)r * HGH + tid] / den[sd * HGH + tid];
  __syncthreads();
  atomicAdd(&xg[(size_t)sd * DDIM + tid],
            alpha[tid >> 6] * xh[(size_t)ss * DDIM + tid]);
}

__global__ void xg_bias_kernel(float* __restrict__ xg, const float* __restrict__ gat_b_l,
                               bf16_t* __restrict__ xgbf) {
  int idx = blockIdx.x * blockDim.x + threadIdx.x;
  if (idx >= NNODE * DDIM) return;
  float v = xg[idx] + gat_b_l[idx & 511];
  xg[idx] = v;
  xgbf[idx] = f2bf(v);
}

// LayerNorm over D=512: block per row, optional residual add, relu, bf16 copy
__global__ __launch_bounds__(256) void ln_kernel(
    const float* __restrict__ a, const float* __restrict__ addb,
    const float* __restrict__ g, const float* __restrict__ be,
    float* __restrict__ out, bf16_t* __restrict__ outbf, int relu) {
  __shared__ float red[256];
  const int row = blockIdx.x, t = threadIdx.x;
  const size_t base = (size_t)row * DDIM;
  float v0 = a[base + t], v1 = a[base + t + 256];
  if (addb) { v0 += addb[base + t]; v1 += addb[base + t + 256]; }
  red[t] = v0 + v1;
  __syncthreads();
  for (int s = 128; s > 0; s >>= 1) { if (t < s) red[t] += red[t + s]; __syncthreads(); }
  float mu = red[0] * (1.f / DDIM);
  __syncthreads();
  float d0 = v0 - mu, d1 = v1 - mu;
  red[t] = d0 * d0 + d1 * d1;
  __syncthreads();
  for (int s = 128; s > 0; s >>= 1) { if (t < s) red[t] += red[t + s]; __syncthreads(); }
  float inv = rsqrtf(red[0] * (1.f / DDIM) + 1e-5f);
  float o0 = d0 * inv * g[t] + be[t];
  float o1 = d1 * inv * g[t + 256] + be[t + 256];
  if (relu) { o0 = fmaxf(o0, 0.f); o1 = fmaxf(o1, 0.f); }
  out[base + t] = o0;
  out[base + t + 256] = o1;
  if (outbf) { outbf[base + t] = f2bf(o0); outbf[base + t + 256] = f2bf(o1); }
}

__global__ void pool_kernel(const float* __restrict__ x, float* __restrict__ pooled) {
  int idx = blockIdx.x * blockDim.x + threadIdx.x;
  if (idx >= BGR * DDIM) return;
  int b = idx >> 9, d = idx & 511;
  float s = 0.f;
  for (int p = 0; p < PNODE; p++) s += x[((size_t)(b * PNODE + p)) * DDIM + d];
  pooled[idx] = s * (1.f / PNODE);
}

__global__ void head_kernel(const float* __restrict__ pooled,
                            const float* __restrict__ W, const float* __restrict__ bb,
                            float* __restrict__ out) {
  int idx = blockIdx.x * blockDim.x + threadIdx.x;
  if (idx >= BGR * OUTD) return;
  int b = idx >> 7, oo = idx & 127;
  float s = bb[oo];
  const float* pr = pooled + (size_t)b * DDIM;
  for (int d = 0; d < DDIM; d++) s += pr[d] * W[(size_t)d * OUTD + oo];
  out[idx] = s;
}

// ---------- host ----------
static inline int cdiv(long long a, long long b) { return (int)((a + b - 1) / b); }

extern "C" void kernel_launch(void* const* d_in, const int* in_sizes, int n_in,
                              void* d_out, int out_size, void* d_ws, size_t ws_size,
                              hipStream_t stream) {
  (void)in_sizes; (void)n_in; (void)out_size; (void)ws_size;
  const int* x_feat   = (const int*)d_in[0];
  const int* eaf      = (const int*)d_in[1];
  const int* eidx     = (const int*)d_in[2];
  const int* src = eidx;
  const int* dst = eidx + EEDGE;
  // d_in[3] = batch (implied by n/128)
  const float* atom_emb  = (const float*)d_in[4];
  const float* bond_emb  = (const float*)d_in[5];
  const float* gat_W     = (const float*)d_in[6];
  const float* gat_We    = (const float*)d_in[7];
  const float* att_src   = (const float*)d_in[8];
  const float* att_dst   = (const float*)d_in[9];
  const float* att_edge  = (const float*)d_in[10];
  const float* gat_b     = (const float*)d_in[11];
  const float* mid_W     = (const float*)d_in[12];
  const float* mid_b     = (const float*)d_in[13];
  const float* mid_ln_g  = (const float*)d_in[14];
  const float* mid_ln_b  = (const float*)d_in[15];
  const float* in_proj_W = (const float*)d_in[16];
  const float* in_proj_b = (const float*)d_in[17];
  const float* out_proj_W= (const float*)d_in[18];
  const float* out_proj_b= (const float*)d_in[19];
  const float* ln1_g     = (const float*)d_in[20];
  const float* ln1_b     = (const float*)d_in[21];
  const float* ln2_g     = (const float*)d_in[22];
  const float* ln2_b     = (const float*)d_in[23];
  const float* ff1_W     = (const float*)d_in[24];
  const float* ff1_b     = (const float*)d_in[25];
  const float* ff2_W     = (const float*)d_in[26];
  const float* ff2_b     = (const float*)d_in[27];
  const float* lin2_W    = (const float*)d_in[28];
  const float* lin2_b    = (const float*)d_in[29];
  float* out = (float*)d_out;

  // workspace carve-out
  char* wp = (char*)d_ws;
  auto alloc = [&](size_t bytes) -> void* {
    void* r = (void*)wp;
    wp += (bytes + 255) & ~(size_t)255;
    return r;
  };
  const size_t ND = (size_t)NNODE * DDIM;
  bf16_t* wb_gat = (bf16_t*)alloc(LLAYER * (size_t)DDIM * DDIM * 2);
  bf16_t* wb_mid = (bf16_t*)alloc(LLAYER * (size_t)DDIM * DDIM * 2);
  bf16_t* wb_in  = (bf16_t*)alloc(LLAYER * (size_t)DDIM * 3 * DDIM * 2);
  bf16_t* wb_out = (bf16_t*)alloc(LLAYER * (size_t)DDIM * DDIM * 2);
  bf16_t* wb_ff1 = (bf16_t*)alloc(LLAYER * (size_t)DDIM * 2 * DDIM * 2);
  bf16_t* wb_ff2 = (bf16_t*)alloc(LLAYER * (size_t)2 * DDIM * DDIM * 2);
  float* weff    = (float*)alloc(32 * DDIM * 4);
  float* mean_e  = (float*)alloc(ND * 4);
  float* deg     = (float*)alloc(NNODE * 4);
  float* a_e_all = (float*)alloc((size_t)ETOT * 32 * 4);
  float*  xf32   = (float*)alloc(ND * 4);
  bf16_t* xbf    = (bf16_t*)alloc(ND * 2);
  float*  xh     = (float*)alloc(ND * 4);
  float*  xg     = (float*)alloc(ND * 4);
  bf16_t* xgbf   = (bf16_t*)alloc(ND * 2);
  float*  tmp    = (float*)alloc(ND * 4);
  float*  x1     = (float*)alloc(ND * 4);
  float*  qkv    = (float*)alloc(ND * 3 * 4);
  float*  obuf   = (float*)alloc(ND * 4);
  bf16_t* obf    = (bf16_t*)alloc(ND * 2);
  float*  x2     = (float*)alloc(ND * 4);
  float*  xc     = (float*)alloc(ND * 4);
  bf16_t* xcbf   = (bf16_t*)alloc(ND * 2);
  float*  ffh    = (float*)alloc(ND * 2 * 4);
  bf16_t* ffhbf  = (bf16_t*)alloc(ND * 2 * 2);
  float*  ff2o   = (float*)alloc(ND * 4);
  float*  a_s    = (float*)alloc((size_t)NNODE * HGH * 4);
  float*  a_d    = (float*)alloc((size_t)NNODE * HGH * 4);
  float*  lg     = (float*)alloc((size_t)ETOT * HGH * 4);
  unsigned* mmax = (unsigned*)alloc((size_t)NNODE * HGH * 4);
  float*  den    = (float*)alloc((size_t)NNODE * HGH * 4);
  float*  pooled = (float*)alloc((size_t)BGR * DDIM * 4);

  // ---- setup: weight bf16 conversion ----
  auto cvt = [&](const float* in, bf16_t* o_, size_t n) {
    cvt_bf16_kernel<<<cdiv(n, 256), 256, 0, stream>>>(in, o_, n);
  };
  cvt(gat_W, wb_gat, LLAYER * (size_t)DDIM * DDIM);
  cvt(mid_W, wb_mid, LLAYER * (size_t)DDIM * DDIM);
  cvt(in_proj_W, wb_in, LLAYER * (size_t)DDIM * 3 * DDIM);
  cvt(out_proj_W, wb_out, LLAYER * (size_t)DDIM * DDIM);
  cvt(ff1_W, wb_ff1, LLAYER * (size_t)DDIM * 2 * DDIM);
  cvt(ff2_W, wb_ff2, LLAYER * (size_t)2 * DDIM * DDIM);
  weff_kernel<<<cdiv(LLAYER * HGH * DDIM, 256), 256, 0, stream>>>(gat_We, att_edge, weff);

  // ---- embeddings, degree, mean edge attr, fused a_e for all layers ----
  (void)hipMemsetAsync(mean_e, 0, ND * 4, stream);
  (void)hipMemsetAsync(deg, 0, NNODE * 4, stream);
  atom_embed_kernel<<<cdiv(ND, 256), 256, 0, stream>>>(x_feat, atom_emb, xf32, xbf);
  deg_kernel<<<cdiv(EEDGE, 256), 256, 0, stream>>>(dst, deg);
  bond_mean_kernel<<<cdiv((size_t)EEDGE * DDIM, 256), 256, 0, stream>>>(eaf, dst,
                                                                        bond_emb, mean_e);
  mean_fin_kernel<<<cdiv(ND, 256), 256, 0, stream>>>(mean_e, deg);
  aeall_kernel<<<ETOT, 256, 0, stream>>>(eaf, bond_emb, mean_e, weff, a_e_all);

  const dim3 gemmB(256);
  for (int l = 0; l < LLAYER; l++) {
    (void)hipMemsetAsync(xg, 0, ND * 4, stream);
    (void)hipMemsetAsync(mmax, 0, (size_t)NNODE * HGH * 4, stream);
    (void)hipMemsetAsync(den, 0, (size_t)NNODE * HGH * 4, stream);

    // xh = x @ gat_W[l]
    gemm_bf16_kernel<<<dim3(DDIM / 128, NNODE / 128), gemmB, 0, stream>>>(
        xbf, wb_gat + (size_t)l * DDIM * DDIM, nullptr, xh, nullptr,
        NNODE, DDIM, DDIM, 0);
    asd_kernel<<<cdiv(NNODE * HGH, 256), 256, 0, stream>>>(
        xh, att_src + (size_t)l * HGH * CGD, att_dst + (size_t)l * HGH * CGD, a_s, a_d);
    logits_kernel<<<cdiv(ETOT * HGH, 256), 256, 0, stream>>>(src, dst, a_s, a_d,
                                                             a_e_all, l, lg, mmax);
    pexp_kernel<<<cdiv(ETOT * HGH, 256), 256, 0, stream>>>(dst, lg, mmax, den);
    agg_kernel<<<ETOT, 512, 0, stream>>>(src, dst, lg, den, xh, xg);
    xg_bias_kernel<<<cdiv(ND, 256), 256, 0, stream>>>(xg, gat_b + (size_t)l * DDIM, xgbf);
    // tmp = xg @ mid_W + mid_b ; x1 = relu(LN(tmp))
    gemm_bf16_kernel<<<dim3(DDIM / 128, NNODE / 128), gemmB, 0, stream>>>(
        xgbf, wb_mid + (size_t)l * DDIM * DDIM, mid_b + (size_t)l * DDIM, tmp, nullptr,
        NNODE, DDIM, DDIM, 0);
    ln_kernel<<<NNODE, 256, 0, stream>>>(tmp, nullptr, mid_ln_g + (size_t)l * DDIM,
                                         mid_ln_b + (size_t)l * DDIM, x1, nullptr, 1);
    // qkv = x @ in_proj_W + b
    gemm_bf16_kernel<<<dim3(3 * DDIM / 128, NNODE / 128), gemmB, 0, stream>>>(
        xbf, wb_in + (size_t)l * DDIM * 3 * DDIM, in_proj_b + (size_t)l * 3 * DDIM,
        qkv, nullptr, NNODE, 3 * DDIM, DDIM, 0);
    attn_kernel<<<dim3(BGR, HGH, PNODE / 32), 256, 0, stream>>>(qkv, obuf, obf);
    // x2 = o @ out_proj_W + b
    gemm_bf16_kernel<<<dim3(DDIM / 128, NNODE / 128), gemmB, 0, stream>>>(
        obf, wb_out + (size_t)l * DDIM * DDIM, out_proj_b + (size_t)l * DDIM, x2, nullptr,
        NNODE, DDIM, DDIM, 0);
    // xc = LN1(x1 + x2)
    ln_kernel<<<NNODE, 256, 0, stream>>>(x1, x2, ln1_g + (size_t)l * DDIM,
                                         ln1_b + (size_t)l * DDIM, xc, xcbf, 0);
    // ffh = relu(xc @ ff1 + b)
    gemm_bf16_kernel<<<dim3(2 * DDIM / 128, NNODE / 128), gemmB, 0, stream>>>(
        xcbf, wb_ff1 + (size_t)l * DDIM * 2 * DDIM, ff1_b + (size_t)l * 2 * DDIM,
        ffh, ffhbf, NNODE, 2 * DDIM, DDIM, 1);
    // ff2o = ffh @ ff2 + b
    gemm_bf16_kernel<<<dim3(DDIM / 128, NNODE / 128), gemmB, 0, stream>>>(
        ffhbf, wb_ff2 + (size_t)l * 2 * DDIM * DDIM, ff2_b + (size_t)l * DDIM,
        ff2o, nullptr, NNODE, DDIM, 2 * DDIM, 0);
    // x = LN2(ff2o + xc)
    ln_kernel<<<NNODE, 256, 0, stream>>>(ff2o, xc, ln2_g + (size_t)l * DDIM,
                                         ln2_b + (size_t)l * DDIM, xf32, xbf, 0);
  }

  pool_kernel<<<cdiv(BGR * DDIM, 256), 256, 0, stream>>>(xf32, pooled);
  head_kernel<<<cdiv(BGR * OUTD, 256), 256, 0, stream>>>(pooled, lin2_W, lin2_b, out);
}
